// MoELayer_2551210574648
// MI455X (gfx1250) — matrix operand order repaired and checked by reading, not compiled, not run on hardware
//
#include <hip/hip_runtime.h>
#include <hip/hip_bf16.h>

#define N_EMBD 768
#define N_EXP  8
#define D_FF   3072
#define N_TOK  16384

#define MT        32      // tokens per workgroup tile
#define FFB       128     // ff chunk
#define XS_STRIDE 776     // 768 + 8 bf16 pad (16B-aligned rows, bank staggering)
#define HS_STRIDE 136     // 128 + 8 bf16 pad

#define XS_BYTES  (MT * XS_STRIDE * 2)            // 49664
#define HS_BYTES  (2 * MT * HS_STRIDE * 2)        // 17408 (double buffered)
#define SMEM_BYTES (XS_BYTES + HS_BYTES)          // 67072

typedef __attribute__((ext_vector_type(16))) __bf16 v16bf;
typedef __attribute__((ext_vector_type(8)))  float  v8f;
typedef __attribute__((ext_vector_type(4)))  int    v4i;

union Frag {
    v16bf v;
    uint4 q[2];
};

__device__ inline v8f zero8() {
    v8f z;
#pragma unroll
    for (int i = 0; i < 8; i++) z[i] = 0.0f;
    return z;
}

// B fragment: per-lane contiguous 16 bf16 starting at p (16B-aligned runs)
__device__ inline v16bf load_frag16(const __bf16* p) {
    Frag f;
    f.q[0] = *(const uint4*)(p);
    f.q[1] = *(const uint4*)(p + 8);
    return f.v;
}

// A fragment from a row pointer: contiguous k = 16*half + s (matches B)
__device__ inline v16bf load_frag_a(const __bf16* rowp, int k0, int half) {
    Frag f;
    f.q[0] = *(const uint4*)(rowp + k0 + half * 16);
    f.q[1] = *(const uint4*)(rowp + k0 + half * 16 + 8);
    return f.v;
}

__device__ inline v8f wmma_bf16(v16bf a, v16bf b, v8f c) {
    return __builtin_amdgcn_wmma_f32_16x16x32_bf16(
        /*neg_a=*/false, a, /*neg_b=*/false, b,
        /*c_mod=*/(short)0, c, /*reuse_a=*/false, /*reuse_b=*/false);
}

// ---- CDNA5 async global->LDS copy (ASYNCcnt-tracked), with asm fallback ----
__device__ inline void async_copy_b128(const __bf16* gp, void* lp) {
#if __has_builtin(__builtin_amdgcn_global_load_async_to_lds_b128)
    __builtin_amdgcn_global_load_async_to_lds_b128(
        (__attribute__((address_space(1))) v4i*)(uintptr_t)gp,
        (__attribute__((address_space(3))) v4i*)lp,
        0, 0);
#else
    uint32_t ldsoff = (uint32_t)(uintptr_t)lp;
    uint64_t gaddr  = (uint64_t)(uintptr_t)gp;
    asm volatile("global_load_async_to_lds_b128 %0, %1, off"
                 :: "v"(ldsoff), "v"(gaddr) : "memory");
#endif
}

__device__ inline void wait_async0() {
#if __has_builtin(__builtin_amdgcn_s_wait_asynccnt)
    __builtin_amdgcn_s_wait_asynccnt(0);
#else
    asm volatile("s_wait_asynccnt 0" ::: "memory");
#endif
}

// ---------------------------------------------------------------------------
// 0) zero per-expert counters
__global__ void k_init(int* cnt) {
    if (threadIdx.x < N_EXP) cnt[threadIdx.x] = 0;
}

// 1) x f32 -> bf16 (coalesced, 4 elems/thread)
__global__ void k_xbf16(const float* __restrict__ x, __bf16* __restrict__ xb) {
    int i = blockIdx.x * blockDim.x + threadIdx.x;      // per 4 elems
    float4 v = ((const float4*)x)[i];
    union { __bf16 h[4]; uint2 u; } o;
    o.h[0] = (__bf16)v.x; o.h[1] = (__bf16)v.y;
    o.h[2] = (__bf16)v.z; o.h[3] = (__bf16)v.w;
    ((uint2*)xb)[i] = o.u;
}

// 2) transpose + f32->bf16: in [E][R][C] -> out [E][C][R]
__global__ void k_transpose_bf16(const float* __restrict__ in, __bf16* __restrict__ out,
                                 int R, int C) {
    long i = (long)blockIdx.x * blockDim.x + threadIdx.x;
    long rc = (long)R * C;
    long total = (long)N_EXP * rc;
    if (i >= total) return;
    int e = (int)(i / rc);
    long rem = i - (long)e * rc;
    int c = (int)(rem / R);
    int r = (int)(rem - (long)c * R);
    out[i] = (__bf16)in[(long)e * rc + (long)r * C + c];
}

// 3) router: logits, top-2, softmax, build per-expert token lists
__global__ void k_router(const float* __restrict__ x, const float* __restrict__ Wr,
                         const float* __restrict__ br, float* __restrict__ wslot,
                         int* __restrict__ tlist, int* __restrict__ cnt) {
    int t = blockIdx.x * blockDim.x + threadIdx.x;
    if (t >= N_TOK) return;
    float acc[N_EXP];
#pragma unroll
    for (int e = 0; e < N_EXP; e++) acc[e] = br[e];
    const float4* xv = (const float4*)(x + (size_t)t * N_EMBD);
    for (int k4 = 0; k4 < N_EMBD / 4; k4++) {
        float4 xx = xv[k4];
        const float* wr = Wr + (size_t)(k4 * 4) * N_EXP;
#pragma unroll
        for (int e = 0; e < N_EXP; e++) {
            acc[e] += xx.x * wr[e] + xx.y * wr[N_EXP + e] +
                      xx.z * wr[2 * N_EXP + e] + xx.w * wr[3 * N_EXP + e];
        }
    }
    float b = -3.4e38f, s = -3.4e38f;
    int be = 0, se = 0;
#pragma unroll
    for (int e = 0; e < N_EXP; e++) {
        float l = acc[e];
        if (l > b)      { s = b; se = be; b = l; be = e; }
        else if (l > s) { s = l; se = e; }
    }
    float d   = __expf(s - b);
    float inv = 1.0f / (1.0f + d);
    wslot[t * 2 + 0] = inv;
    wslot[t * 2 + 1] = d * inv;
    int p0 = atomicAdd(&cnt[be], 1);
    tlist[be * N_TOK + p0] = (t << 1);
    int p1 = atomicAdd(&cnt[se], 1);
    tlist[se * N_TOK + p1] = (t << 1) | 1;
}

// 4) per-expert fused FFN: h = relu(Xg @ W1 + b1); y = h @ W2 + b2 -> ybuf
__global__ void __launch_bounds__(256)
k_ffn(const __bf16* __restrict__ xb, const __bf16* __restrict__ W1T,
      const float* __restrict__ b1, const __bf16* __restrict__ W2T,
      const float* __restrict__ b2, const int* __restrict__ tlist,
      const int* __restrict__ cnt, __bf16* __restrict__ ybuf) {
    extern __shared__ char smem[];
    __bf16* Xs  = (__bf16*)smem;                 // [MT][XS_STRIDE]
    __bf16* Hs0 = (__bf16*)(smem + XS_BYTES);    // [2][MT][HS_STRIDE]
    __shared__ int rowent[MT];

    const int e = blockIdx.y;
    const int n_e = cnt[e];
    const int base = blockIdx.x * MT;
    if (base >= n_e) return;

    const int tid  = threadIdx.x;
    const int lane = tid & 31;
    const int wv   = tid >> 5;          // 8 waves
    const int half = lane >> 4;
    const int lc   = lane & 15;

    if (tid < MT) {
        rowent[tid] = (base + tid < n_e) ? tlist[e * N_TOK + base + tid] : -1;
    }

    // ---- async gather of the X tile (bf16) straight into LDS ----
    // tile = MT rows x 1536B; 256 lanes x 16B = 4096B per pass, 12 passes.
    {
#pragma unroll
        for (int p = 0; p < (MT * N_EMBD * 2) / (256 * 16); p++) {
            int lb  = p * 4096 + tid * 16;       // byte position within tile
            int row = lb / (N_EMBD * 2);
            int col = lb - row * (N_EMBD * 2);   // bytes within row
            int li  = base + row;
            if (li >= n_e) li = n_e - 1;         // clamp: garbage rows are never stored
            int tok = tlist[e * N_TOK + li] >> 1;
            const __bf16* gp = xb + (size_t)tok * N_EMBD + (col >> 1);
            async_copy_b128(gp, (char*)(Xs + row * XS_STRIDE) + col);
        }
        wait_async0();
    }
    __syncthreads();

    const __bf16* W1e = W1T + (size_t)e * D_FF * N_EMBD;   // [3072][768]
    const __bf16* W2e = W2T + (size_t)e * N_EMBD * D_FF;   // [768][3072]

    v8f y[6][2];
#pragma unroll
    for (int j = 0; j < 6; j++) { y[j][0] = zero8(); y[j][1] = zero8(); }

    for (int cc = 0; cc < D_FF / FFB; cc++) {
        const int ffbase = cc * FFB;
        __bf16* Hc = Hs0 + (cc & 1) * (MT * HS_STRIDE);    // double buffer

        // ---- GEMM1: this wave computes h[:, wv*16 .. wv*16+15] ----
        v8f h0 = zero8(), h1 = zero8();
        const __bf16* w1p = W1e + (size_t)(ffbase + wv * 16 + lc) * N_EMBD;
#pragma unroll 4
        for (int k0 = 0; k0 < N_EMBD; k0 += 32) {
            v16bf B  = load_frag16(w1p + k0 + half * 16);
            v16bf A0 = load_frag_a(&Xs[lc * XS_STRIDE], k0, half);
            v16bf A1 = load_frag_a(&Xs[(16 + lc) * XS_STRIDE], k0, half);
            h0 = wmma_bf16(A0, B, h0);
            h1 = wmma_bf16(A1, B, h1);
        }
        float b1v = b1[e * D_FF + ffbase + wv * 16 + lc];
#pragma unroll
        for (int i = 0; i < 8; i++) {
            float v0 = h0[i] + b1v; v0 = v0 > 0.f ? v0 : 0.f;
            float v1 = h1[i] + b1v; v1 = v1 > 0.f ? v1 : 0.f;
            Hc[(half * 8 + i) * HS_STRIDE + wv * 16 + lc]      = (__bf16)v0;
            Hc[(16 + half * 8 + i) * HS_STRIDE + wv * 16 + lc] = (__bf16)v1;
        }
        __syncthreads();   // single barrier per chunk (double-buffered H)

        // ---- GEMM2: y[:, wv*96 .. wv*96+95] += Hchunk @ W2chunk ----
#pragma unroll
        for (int k0 = 0; k0 < FFB; k0 += 32) {
            v16bf A0 = load_frag_a(&Hc[lc * HS_STRIDE], k0, half);
            v16bf A1 = load_frag_a(&Hc[(16 + lc) * HS_STRIDE], k0, half);
#pragma unroll
            for (int j = 0; j < 6; j++) {
                const __bf16* bp = W2e + (size_t)(wv * 96 + j * 16 + lc) * D_FF
                                   + ffbase + k0 + half * 16;
                v16bf B = load_frag16(bp);
                y[j][0] = wmma_bf16(A0, B, y[j][0]);
                y[j][1] = wmma_bf16(A1, B, y[j][1]);
            }
        }
    }

    // ---- epilogue: +b2, scatter bf16 into ybuf[(tok*2+slot)][col] ----
#pragma unroll
    for (int j = 0; j < 6; j++) {
        int col = wv * 96 + j * 16 + lc;
        float b2v = b2[e * N_EMBD + col];
#pragma unroll
        for (int mt = 0; mt < 2; mt++) {
#pragma unroll
            for (int i = 0; i < 8; i++) {
                int row = mt * 16 + half * 8 + i;
                int ent = rowent[row];
                if (ent >= 0) {
                    float v = y[j][mt][i] + b2v;
                    ybuf[(size_t)ent * N_EMBD + col] = (__bf16)v;
                }
            }
        }
    }
}

// 5) combine: out[t] = w0*y[t,slot0] + w1*y[t,slot1]
__global__ void k_combine(const __bf16* __restrict__ ybuf, const float* __restrict__ wslot,
                          float* __restrict__ out) {
    int i = blockIdx.x * blockDim.x + threadIdx.x;
    if (i >= N_TOK * N_EMBD) return;
    int t = i / N_EMBD;
    int c = i - t * N_EMBD;
    float w0 = wslot[t * 2 + 0];
    float w1 = wslot[t * 2 + 1];
    float y0 = (float)ybuf[(size_t)(t * 2 + 0) * N_EMBD + c];
    float y1 = (float)ybuf[(size_t)(t * 2 + 1) * N_EMBD + c];
    out[i] = w0 * y0 + w1 * y1;
}

extern "C" void kernel_launch(void* const* d_in, const int* in_sizes, int n_in,
                              void* d_out, int out_size, void* d_ws, size_t ws_size,
                              hipStream_t stream) {
    const float* x  = (const float*)d_in[0];
    const float* Wr = (const float*)d_in[1];
    const float* br = (const float*)d_in[2];
    const float* W1 = (const float*)d_in[3];
    const float* b1 = (const float*)d_in[4];
    const float* W2 = (const float*)d_in[5];
    const float* b2 = (const float*)d_in[6];
    float* out = (float*)d_out;

    char* ws = (char*)d_ws;
    const size_t szWT = (size_t)N_EXP * N_EMBD * D_FF * sizeof(__bf16); // 37,748,736
    const size_t szYB = (size_t)N_TOK * 2 * N_EMBD * sizeof(__bf16);    // 50,331,648
    const size_t szXB = (size_t)N_TOK * N_EMBD * sizeof(__bf16);        // 25,165,824
    __bf16* W1T  = (__bf16*)(ws);
    __bf16* W2T  = (__bf16*)(ws + szWT);
    __bf16* ybuf = (__bf16*)(ws + 2 * szWT);
    __bf16* xbuf = (__bf16*)(ws + 2 * szWT + szYB);
    float*  wslot = (float*)(ws + 2 * szWT + szYB + szXB);
    int*    cnt   = (int*)((char*)wslot + (size_t)N_TOK * 2 * sizeof(float));
    int*    tlist = (int*)((char*)cnt + 256);

    k_init<<<1, 32, 0, stream>>>(cnt);

    k_xbf16<<<(N_TOK * N_EMBD / 4) / 256, 256, 0, stream>>>(x, xbuf);

    const int tpBlocks = (N_EXP * N_EMBD * D_FF) / 256;        // 73728
    k_transpose_bf16<<<tpBlocks, 256, 0, stream>>>(W1, W1T, N_EMBD, D_FF);
    k_transpose_bf16<<<tpBlocks, 256, 0, stream>>>(W2, W2T, D_FF, N_EMBD);

    k_router<<<N_TOK / 256, 256, 0, stream>>>(x, Wr, br, wslot, tlist, cnt);

    dim3 ffnGrid(N_TOK / MT, N_EXP);                           // 512 x 8
    k_ffn<<<ffnGrid, 256, SMEM_BYTES, stream>>>(xbuf, W1T, b1, W2T, b2, tlist, cnt, ybuf);

    k_combine<<<(N_TOK * N_EMBD) / 256, 256, 0, stream>>>(ybuf, wslot, out);
}
